// LongitudE_48447231099018
// MI455X (gfx1250) — matrix operand order repaired and checked by reading, not compiled, not run on hardware
//
#include <hip/hip_runtime.h>
#include <math.h>

// ---------------- types ----------------
typedef __attribute__((ext_vector_type(16))) __bf16 v16bf;
typedef __attribute__((ext_vector_type(2)))  __bf16 v2bf;
typedef __attribute__((ext_vector_type(8)))  float  v8f;
typedef __attribute__((ext_vector_type(4)))  unsigned int u32x4;
typedef __attribute__((ext_vector_type(4)))  int i32x4;
typedef __attribute__((ext_vector_type(8)))  int i32x8;

union FragBF {          // 32 bytes = 16 bf16 = one WMMA 16x16x32 bf16 operand per lane
    v16bf v;
    uint4 u[2];
    v2bf  h[8];
};

// ---------------- problem constants ----------------
#define NB_     2
#define B_      256
#define DIM_    400
#define CD_     400          // TuckER core dim
#define HD_     200
#define ROWS_   1024         // NB * 2B rows of the big contraction
#define DPAD_   416          // K padded to 13 chunks of 32
#define EPAD_   416          // N padded to 26 tiles of 16
#define KSPLIT_ 16
#define RPK_    25           // 400 / KSPLIT r-values per split
#define HPITCH_ 424          // hn row pitch (bf16) in BOTH global and LDS: 212 dwords == 20 mod 64 banks
#define WRB_    (416 * 416 * 2)   // bytes per Wt[r] slab = 346112
#define NTB_    (2 * 16 * 416 * 2) // byte stride between a wave's consecutive N-tiles = 26624

#define PI_     3.14159265358979323846f
#define HPI_    1.57079632679489661923f
#define ASCALE_ (PI_ * 400.0f / 26.0f)   // pi / EMB_RANGE
#define BNEPS_  1e-5f

// ---------------- workspace layout (bytes) ----------------
#define SZ_WT    ((size_t)400 * EPAD_ * DPAD_ * 2)        // bf16 Wt[r][e][d]
#define OFF_WT   ((size_t)0)
#define OFF_HNP  (OFF_WT + SZ_WT)                          // bf16 hn [1024][424]
#define SZ_HNP   ((size_t)ROWS_ * HPITCH_ * 2)
#define OFF_SA   (OFF_HNP + SZ_HNP)                        // f32 src_axis [2][256][400]
#define SZ_SA    ((size_t)NB_ * B_ * DIM_ * 4)
#define OFF_RR   (OFF_SA + SZ_SA)                          // f32 rr [1024][400]
#define SZ_RR    ((size_t)ROWS_ * CD_ * 4)
#define OFF_MU0  (OFF_RR + SZ_RR)
#define OFF_RS0  (OFF_MU0 + 4096)
#define OFF_MU1  (OFF_RS0 + 4096)
#define OFF_RS1  (OFF_MU1 + 4096)
#define OFF_TP   (OFF_RS1 + 4096)                          // f32 tpart [16][1024][400]
#define SZ_TP    ((size_t)KSPLIT_ * ROWS_ * CD_ * 4)
#define OFF_T    (OFF_TP + SZ_TP)                          // f32 t [1024][400]
#define SZ_T     ((size_t)ROWS_ * CD_ * 4)
#define OFF_AXE  (OFF_T + SZ_T)                            // f32 axis_e [2][256][400]
#define OFF_AGE  (OFF_AXE + SZ_SA)                         // f32 arg_e  [2][256][400]
#define OFF_LOG  (OFF_AGE + SZ_SA)                         // f32 logits [2][256][2][400]
#define SZ_LOG   ((size_t)NB_ * B_ * 2 * 400 * 4)
#define OFF_A1   (OFF_LOG + SZ_LOG)                        // f32 a1  [1024][200]
#define SZ_A1    ((size_t)ROWS_ * HD_ * 4)
#define OFF_RG1  (OFF_A1 + SZ_A1)                          // f32 rg1 [1024][200]

__device__ __forceinline__ unsigned short f2bf(float f) {
    union { float f; unsigned u; } c; c.f = f;
    unsigned u = c.u;
    unsigned r = (u + 0x7FFFu + ((u >> 16) & 1u)) >> 16;   // round-to-nearest-even
    return (unsigned short)r;
}

// ============ K0: gather + angle transforms -> src_axis (f32), rr rows (f32) ============
__global__ __launch_bounds__(256) void k_embed(const int* __restrict__ eidx,
                                               const int* __restrict__ ridx,
                                               const float* __restrict__ ee,
                                               const float* __restrict__ ra,
                                               const float* __restrict__ rg,
                                               float* __restrict__ sa,
                                               float* __restrict__ rr) {
    int nb = blockIdx.x;               // 0..511
    int n = nb / B_, b = nb % B_;
    int ei = eidx[n * B_ + b];
    int rx = ridx[n * B_ + b];
    for (int c = threadIdx.x; c < DIM_; c += 256) {
        sa[((size_t)n * B_ + b) * DIM_ + c] = tanhf(ASCALE_ * ee[(size_t)ei * DIM_ + c]) * PI_;
        float rax = tanhf(ASCALE_ * ra[(size_t)rx * DIM_ + c]) * PI_;
        float rag = tanhf(2.0f * ASCALE_ * rg[(size_t)rx * DIM_ + c]) * HPI_ + HPI_;
        size_t p0 = ((size_t)n * 512 + 2 * b) * CD_ + c;
        rr[p0] = rax;
        rr[p0 + CD_] = rag;
    }
}

// ============ K1: BN0 batch stats per (n,c) over 512 rows (odd rows are zero) ============
__global__ __launch_bounds__(256) void k_bn0stats(const float* __restrict__ sa,
                                                  float* __restrict__ mu, float* __restrict__ rs) {
    int f = blockIdx.x;                 // n*400 + c
    int n = f / CD_, c = f % CD_;
    __shared__ float ss[256], sq[256];
    float x = sa[((size_t)n * B_ + threadIdx.x) * DIM_ + c];
    ss[threadIdx.x] = x; sq[threadIdx.x] = x * x;
    __syncthreads();
    for (int s = 128; s > 0; s >>= 1) {
        if (threadIdx.x < s) { ss[threadIdx.x] += ss[threadIdx.x + s]; sq[threadIdx.x] += sq[threadIdx.x + s]; }
        __syncthreads();
    }
    if (threadIdx.x == 0) {
        float m = ss[0] / 512.0f;                 // 256 real rows + 256 zero rows
        float q = sq[0] / 512.0f;
        mu[f] = m;
        rs[f] = rsqrtf(q - m * m + BNEPS_);
    }
}

// ============ K2: apply BN0, emit bf16 hn [1024][424] (pitch matches LDS) ============
__global__ __launch_bounds__(256) void k_makehn(const float* __restrict__ sa,
                                                const float* __restrict__ mu, const float* __restrict__ rs,
                                                const float* __restrict__ g0, const float* __restrict__ be0,
                                                unsigned short* __restrict__ hnp) {
    int p = blockIdx.x;                 // 0..1023
    int n = p >> 9, q = p & 511, b = q >> 1, half = q & 1;
    for (int c = threadIdx.x; c < HPITCH_; c += 256) {
        float hv = 0.0f;
        if (c < CD_) {
            float val = half ? 0.0f : sa[((size_t)n * B_ + b) * DIM_ + c];
            int f = n * CD_ + c;
            hv = g0[c] * (val - mu[f]) * rs[f] + be0[c];
        }
        hnp[(size_t)p * HPITCH_ + c] = f2bf(hv);
    }
}

// ============ K3: W fp32 [r][d][e] -> bf16 Wt [r][e(416)][d(416)], LDS 32x33 transpose ============
__global__ __launch_bounds__(256) void k_convW(const float* __restrict__ W,
                                               unsigned short* __restrict__ Wt) {
    int r = blockIdx.x, db = blockIdx.y, eb = blockIdx.z;   // 400 x 13 x 13
    __shared__ float tile[32][33];
    #pragma unroll
    for (int j = 0; j < 4; ++j) {
        int lin = threadIdx.x + j * 256;
        int dd = lin >> 5, e0 = lin & 31;
        int d = db * 32 + dd, e = eb * 32 + e0;
        float v = (d < CD_ && e < CD_) ? W[((size_t)r * CD_ + d) * CD_ + e] : 0.0f;
        tile[dd][e0] = v;
    }
    __syncthreads();
    #pragma unroll
    for (int j = 0; j < 4; ++j) {
        int lin = threadIdx.x + j * 256;
        int e0 = lin >> 5, dd = lin & 31;
        int d = db * 32 + dd, e = eb * 32 + e0;
        Wt[((size_t)r * EPAD_ + e) * DPAD_ + d] = f2bf(tile[dd][e0]);
    }
}

// ============ K4: the big contraction as bf16 WMMA GEMM ============
// t[p,e] = sum_r rr[p,r] * (hn[p,:] @ W[r,:,:])[e]
// grid (16 M-blocks of 64 rows, 16 K-splits of 25 r's); 8 waves = 4 M-subtiles x 2 N-columns.
// hn slab staged to LDS via the Tensor Data Mover; all B loads are one base + 24-bit immediates.
// B fragments: 4-deep rotating software pipeline; A fragments: double-buffered LDS loads.
#define LOADB(f, t, dc) do {                                                   \
    const uint4* bp_ = (const uint4*)(wbase + (t) * NTB_ + (dc) * 64);         \
    (f).u[0] = bp_[0]; (f).u[1] = bp_[2]; } while (0)

__global__ __launch_bounds__(256) void k_gemm(const unsigned short* __restrict__ hnp,
                                              const float* __restrict__ rr,
                                              const unsigned short* __restrict__ Wt,
                                              float* __restrict__ tpart) {
    __shared__ __align__(16) unsigned short sh_hn[64 * HPITCH_];   // 54,272 B
    __shared__ float sh_rr[RPK_ * 64];                             //  6,400 B
    const int mblk = blockIdx.x, ky = blockIdx.y;
    const int tid = threadIdx.x;

    // ---- stage hn slab (64 rows x 424 bf16, contiguous in global at the same pitch) ----
#if __has_builtin(__builtin_amdgcn_tensor_load_to_lds)
    if (tid < 32) {   // one TDM op per workgroup (EXEC-independent, wave-scoped issue)
        unsigned long long ga =
            (unsigned long long)(uintptr_t)(hnp + (size_t)mblk * 64 * HPITCH_);
        unsigned lo = (unsigned)(uintptr_t)(&sh_hn[0]);            // LDS byte offset (low bits)
        u32x4 g0;
        g0[0] = 1u;                                                // count=1, user descriptor
        g0[1] = lo;                                                // lds_addr
        g0[2] = (unsigned)(ga & 0xffffffffull);                    // global_addr[31:0]
        g0[3] = (unsigned)((ga >> 32) & 0x01ffffffull) | 0x80000000u; // addr[56:32] | type=2
        i32x8 g1;
        g1[0] = 0x00010000;                 // data_size = 1 (2-byte elements)
        g1[1] = (int)(HPITCH_ << 16);       // tensor_dim0 = 424
        g1[2] = (int)(1024u << 16);         // tensor_dim1 = 1024 (low 16)
        g1[3] = (int)(HPITCH_ << 16);       // tile_dim0 = 424
        g1[4] = 64;                         // tile_dim1 = 64 rows
        g1[5] = HPITCH_;                    // tensor_dim0_stride = 424
        g1[6] = 0;
        g1[7] = 0;
        i32x4 z4 = {0, 0, 0, 0};
#if defined(__clang_major__) && (__clang_major__ >= 23)
        i32x8 z8 = {0, 0, 0, 0, 0, 0, 0, 0};
        __builtin_amdgcn_tensor_load_to_lds(g0, g1, z4, z4, z8, 0);
#else
        __builtin_amdgcn_tensor_load_to_lds(g0, g1, z4, z4, 0);
#endif
        __builtin_amdgcn_s_wait_tensorcnt(0);
    }
#else
    for (int idx = tid; idx < 64 * 53; idx += 256) {
        int row = idx / 53, c8 = idx % 53;
        *(uint4*)(&sh_hn[row * HPITCH_ + c8 * 8]) =
            *(const uint4*)(&hnp[(size_t)(mblk * 64 + row) * HPITCH_ + c8 * 8]);
    }
#endif
    // ---- stage rr slice [25 r][64 rows] ----
    for (int idx = tid; idx < RPK_ * 64; idx += 256) {
        int ri = idx >> 6, row = idx & 63;
        sh_rr[idx] = rr[(size_t)(mblk * 64 + row) * CD_ + ky * RPK_ + ri];
    }
    __syncthreads();

    const int wave = tid >> 5, lane = tid & 31;
    const int ln = lane & 15, ks = lane >> 4;
    const int mi = wave >> 1, nc = wave & 1;
    const int mrow = mi * 16 + ln;

    v8f acc[13];
    #pragma unroll
    for (int t = 0; t < 13; ++t)
        #pragma unroll
        for (int i = 0; i < 8; ++i) acc[t][i] = 0.0f;

    // single A base (LDS) and single B base (global); everything else is immediates
    const unsigned short* abase = &sh_hn[mrow * HPITCH_ + ks * 8];
    const char* wbase = (const char*)Wt + (size_t)ky * RPK_ * WRB_
                      + (size_t)(nc * 16 + ln) * 832 + (size_t)ks * 16;

    for (int ri = 0; ri < RPK_; ++ri) {
        __builtin_prefetch(wbase + WRB_, 0, 0);            // global_prefetch_b8, next r slab
        const __bf16 sb = (__bf16)sh_rr[ri * 64 + mrow];

        FragBF araw[2];                                    // A raw double buffer across dc
        araw[0].u[0] = *(const uint4*)(abase);
        araw[0].u[1] = *(const uint4*)(abase + 16);

        #pragma unroll
        for (int dc = 0; dc < 13; ++dc) {
            if (dc + 1 < 13) {                             // issue next A loads early (ds latency hidden)
                araw[(dc + 1) & 1].u[0] = *(const uint4*)(abase + (dc + 1) * 32);
                araw[(dc + 1) & 1].u[1] = *(const uint4*)(abase + (dc + 1) * 32 + 16);
            }
            FragBF a;                                      // scale by rr[p,r]: v_pk_mul_bf16
            #pragma unroll
            for (int j = 0; j < 8; ++j) a.h[j] = araw[dc & 1].h[j] * sb;

            FragBF bf[4];                                  // 4-deep B pipeline
            LOADB(bf[0], 0, dc);
            LOADB(bf[1], 1, dc);
            LOADB(bf[2], 2, dc);
            LOADB(bf[3], 3, dc);
            #pragma unroll
            for (int t = 0; t < 13; ++t) {                 // wave's N-tiles: e-tile = nc + 2t
                acc[t] = __builtin_amdgcn_wmma_f32_16x16x32_bf16(
                    false, a.v, false, bf[t & 3].v, (short)0, acc[t], false, false);
                if (t + 4 < 13) LOADB(bf[t & 3], t + 4, dc);  // refill freed slot
            }
        }
        wbase += WRB_;
    }

    // write f32 partials: VGPR i holds (M = i + 8*ks, N = ln)
    float* tp = tpart + ((size_t)ky * ROWS_ + mblk * 64 + mi * 16) * CD_;
    #pragma unroll
    for (int t = 0; t < 13; ++t) {
        const int e0 = (nc + 2 * t) * 16;
        if (e0 < CD_) {
            #pragma unroll
            for (int i = 0; i < 8; ++i)
                tp[(size_t)(i + 8 * ks) * CD_ + e0 + ln] = acc[t][i];
        }
    }
}

// ============ K5: deterministic K-split reduction ============
__global__ __launch_bounds__(256) void k_reduce(const float* __restrict__ tpart, float* __restrict__ t) {
    size_t idx = (size_t)blockIdx.x * 256 + threadIdx.x;   // 1024*400 elements
    if (idx >= (size_t)ROWS_ * CD_) return;
    float s = 0.0f;
    #pragma unroll
    for (int ky = 0; ky < KSPLIT_; ++ky) s += tpart[(size_t)ky * ROWS_ * CD_ + idx];
    t[idx] = s;
}

// ============ K6: BN1 batch stats per (n,c) over 512 rows ============
__global__ __launch_bounds__(256) void k_bn1stats(const float* __restrict__ t,
                                                  float* __restrict__ mu, float* __restrict__ rs) {
    int f = blockIdx.x;
    int n = f / CD_, c = f % CD_;
    __shared__ float ss[256], sq[256];
    float x1 = t[((size_t)n * 512 + threadIdx.x) * CD_ + c];
    float x2 = t[((size_t)n * 512 + threadIdx.x + 256) * CD_ + c];
    ss[threadIdx.x] = x1 + x2; sq[threadIdx.x] = x1 * x1 + x2 * x2;
    __syncthreads();
    for (int s = 128; s > 0; s >>= 1) {
        if (threadIdx.x < s) { ss[threadIdx.x] += ss[threadIdx.x + s]; sq[threadIdx.x] += sq[threadIdx.x + s]; }
        __syncthreads();
    }
    if (threadIdx.x == 0) {
        float m = ss[0] / 512.0f, q = sq[0] / 512.0f;
        mu[f] = m;
        rs[f] = rsqrtf(q - m * m + BNEPS_);
    }
}

// ============ K7: BN1 + cone projection trig -> axis_e, arg_e, logits ============
__global__ __launch_bounds__(256) void k_cone(const float* __restrict__ t,
                                              const float* __restrict__ mu, const float* __restrict__ rs,
                                              const float* __restrict__ g1, const float* __restrict__ be1,
                                              float* __restrict__ axE, float* __restrict__ agE,
                                              float* __restrict__ logits) {
    int nb = blockIdx.x;                // 0..511
    int n = nb / B_, b = nb % B_;
    for (int c = threadIdx.x; c < CD_; c += 256) {
        int f = n * CD_ + c;
        float axp = (t[((size_t)n * 512 + 2 * b) * CD_ + c] - mu[f]) * rs[f] * g1[c] + be1[c];
        float agp = (t[((size_t)n * 512 + 2 * b + 1) * CD_ + c] - mu[f]) * rs[f] * g1[c] + be1[c];
        float axe = tanhf(axp) * PI_;
        float age = tanhf(2.0f * agp) * HPI_ + HPI_;
        axE[((size_t)n * B_ + b) * DIM_ + c] = axe;
        agE[((size_t)n * B_ + b) * DIM_ + c] = age;
        int h = c / HD_, k = c % HD_;
        size_t lb = (((size_t)n * B_ + b) * 2 + h) * 400;
        logits[lb + k]       = axe - 0.5f * age;   // al
        logits[lb + 200 + k] = axe + 0.5f * age;   // ah
    }
}

// ============ K8: first attention layer (both heads: axis & arg), relu ============
__global__ __launch_bounds__(256) void k_fc1(const float* __restrict__ logits,
                                             const float* __restrict__ Wax1, const float* __restrict__ bax1,
                                             const float* __restrict__ Warg1, const float* __restrict__ barg1,
                                             float* __restrict__ a1, float* __restrict__ rg1) {
    int row = blockIdx.x;               // (n*256+b)*2+h, 0..1023
    __shared__ float L[400];
    for (int c = threadIdx.x; c < 400; c += 256) L[c] = logits[(size_t)row * 400 + c];
    __syncthreads();
    int d = threadIdx.x;
    if (d < HD_) {
        float s1 = bax1[d], s2 = barg1[d];
        const float* w1 = Wax1 + (size_t)d * 400;
        const float* w2 = Warg1 + (size_t)d * 400;
        #pragma unroll 4
        for (int k = 0; k < 400; ++k) { s1 += L[k] * w1[k]; s2 += L[k] * w2[k]; }
        a1[(size_t)row * HD_ + d]  = fmaxf(s1, 0.0f);
        rg1[(size_t)row * HD_ + d] = fmaxf(s2, 0.0f);
    }
}

// ============ K9: attention layer 2, softmax over branches, angle/gate combine ============
__global__ __launch_bounds__(256) void k_final(const float* __restrict__ a1, const float* __restrict__ rg1,
                                               const float* __restrict__ Wax2, const float* __restrict__ bax2,
                                               const float* __restrict__ Warg2, const float* __restrict__ barg2,
                                               const float* __restrict__ axE, const float* __restrict__ agE,
                                               float* __restrict__ out) {
    int idx = blockIdx.x;               // b*2+h, 0..511
    int b = idx >> 1, h = idx & 1;
    int row0 = b * 2 + h;               // n=0
    int row1 = 512 + b * 2 + h;         // n=1
    __shared__ float A0[HD_], A1[HD_], G[HD_];
    for (int d = threadIdx.x; d < HD_; d += 256) {
        A0[d] = a1[(size_t)row0 * HD_ + d];
        A1[d] = a1[(size_t)row1 * HD_ + d];
        G[d]  = 0.5f * (rg1[(size_t)row0 * HD_ + d] + rg1[(size_t)row1 * HD_ + d]);
    }
    __syncthreads();
    int d = threadIdx.x;
    if (d < HD_) {
        float z0 = bax2[d], z1 = bax2[d], gz = barg2[d];
        const float* w2 = Wax2 + (size_t)d * HD_;
        const float* wg = Warg2 + (size_t)d * HD_;
        #pragma unroll 4
        for (int k = 0; k < HD_; ++k) { z0 += A0[k] * w2[k]; z1 += A1[k] * w2[k]; gz += G[k] * wg[k]; }
        float m = fmaxf(z0, z1);
        float e0 = expf(z0 - m), e1 = expf(z1 - m);
        float inv = 1.0f / (e0 + e1);
        float at0 = e0 * inv, at1 = e1 * inv;
        int c = h * HD_ + d;
        float ax0 = axE[(size_t)b * DIM_ + c],            ax1 = axE[(size_t)(B_ + b) * DIM_ + c];
        float ag0 = agE[(size_t)b * DIM_ + c],            ag1 = agE[(size_t)(B_ + b) * DIM_ + c];
        float xc = at0 * cosf(ax0) + at1 * cosf(ax1);
        float ys = at0 * sinf(ax0) + at1 * sinf(ax1);
        if (fabsf(xc) < 0.001f) xc = 0.001f;
        float ang = atanf(ys / xc);
        if (xc < 0.0f && ys >= 0.0f) ang += PI_;
        else if (xc < 0.0f && ys < 0.0f) ang -= PI_;
        float gate = 1.0f / (1.0f + expf(-gz));
        float agm = fminf(ag0, ag1) * gate;
        out[(size_t)b * DIM_ + c] = ang;                          // output[0]
        out[(size_t)B_ * DIM_ + (size_t)b * DIM_ + c] = agm;      // output[1]
    }
}

// ============================== launch ==============================
extern "C" void kernel_launch(void* const* d_in, const int* in_sizes, int n_in,
                              void* d_out, int out_size, void* d_ws, size_t ws_size,
                              hipStream_t stream) {
    const int*   eidx = (const int*)d_in[0];
    const int*   ridx = (const int*)d_in[1];
    const float* ee   = (const float*)d_in[2];
    const float* ra   = (const float*)d_in[3];
    const float* rg   = (const float*)d_in[4];
    const float* W    = (const float*)d_in[5];
    const float* g0   = (const float*)d_in[6];
    const float* be0  = (const float*)d_in[7];
    const float* g1   = (const float*)d_in[8];
    const float* be1  = (const float*)d_in[9];
    const float* Wax1 = (const float*)d_in[10];
    const float* bax1 = (const float*)d_in[11];
    const float* Wax2 = (const float*)d_in[12];
    const float* bax2 = (const float*)d_in[13];
    const float* Warg1= (const float*)d_in[14];
    const float* barg1= (const float*)d_in[15];
    const float* Warg2= (const float*)d_in[16];
    const float* barg2= (const float*)d_in[17];

    char* ws = (char*)d_ws;
    unsigned short* Wt  = (unsigned short*)(ws + OFF_WT);
    unsigned short* hnp = (unsigned short*)(ws + OFF_HNP);
    float* sa   = (float*)(ws + OFF_SA);
    float* rr   = (float*)(ws + OFF_RR);
    float* mu0  = (float*)(ws + OFF_MU0);
    float* rs0  = (float*)(ws + OFF_RS0);
    float* mu1  = (float*)(ws + OFF_MU1);
    float* rs1  = (float*)(ws + OFF_RS1);
    float* tpart= (float*)(ws + OFF_TP);
    float* t    = (float*)(ws + OFF_T);
    float* axE  = (float*)(ws + OFF_AXE);
    float* agE  = (float*)(ws + OFF_AGE);
    float* logi = (float*)(ws + OFF_LOG);
    float* a1   = (float*)(ws + OFF_A1);
    float* rg1  = (float*)(ws + OFF_RG1);
    float* out  = (float*)d_out;

    k_embed   <<<dim3(NB_ * B_),      256, 0, stream>>>(eidx, ridx, ee, ra, rg, sa, rr);
    k_bn0stats<<<dim3(NB_ * CD_),     256, 0, stream>>>(sa, mu0, rs0);
    k_makehn  <<<dim3(ROWS_),         256, 0, stream>>>(sa, mu0, rs0, g0, be0, hnp);
    k_convW   <<<dim3(400, 13, 13),   256, 0, stream>>>(W, Wt);
    k_gemm    <<<dim3(16, KSPLIT_),   256, 0, stream>>>(hnp, rr, Wt, tpart);
    k_reduce  <<<dim3((ROWS_ * CD_ + 255) / 256), 256, 0, stream>>>(tpart, t);
    k_bn1stats<<<dim3(NB_ * CD_),     256, 0, stream>>>(t, mu1, rs1);
    k_cone    <<<dim3(NB_ * B_),      256, 0, stream>>>(t, mu1, rs1, g1, be1, axE, agE, logi);
    k_fc1     <<<dim3(ROWS_),         256, 0, stream>>>(logi, Wax1, bax1, Warg1, barg1, a1, rg1);
    k_final   <<<dim3(B_ * 2),        256, 0, stream>>>(a1, rg1, Wax2, bax2, Warg2, barg2, axE, agE, out);
    (void)in_sizes; (void)n_in; (void)out_size; (void)ws_size;
}
